// DecoderGRU_4990751998166
// MI455X (gfx1250) — matrix operand order repaired and checked by reading, not compiled, not run on hardware
//
#include <hip/hip_runtime.h>
#include <stdint.h>

// ---------------------------------------------------------------------------
// DecoderGRU for MI455X (gfx1250, wave32, WMMA).
// GEMMs: v_wmma_f32_16x16x32_bf16, f32 accumulate. Weights converted once to
// bf16 + transposed (104 MB -> L2-resident); 64 sequential steps stream
// weights from L2. Wave tile 32x64; ping-pong (2x unrolled) fragment buffers
// so each WMMA batch overlaps the L2 latency of the other set's loads,
// with no rotate copies and no scratch spills (launch_bounds raises the
// VGPR budget past the ~190 live registers this tile needs).
// ---------------------------------------------------------------------------

typedef __attribute__((ext_vector_type(16))) __bf16 v16bf;
typedef __attribute__((ext_vector_type(8)))  float  v8f;

constexpr int Bn  = 512;
constexpr int Ln  = 256;
constexpr int Dn  = 128;
constexpr int Tn  = 64;
constexpr int Cn  = 12;
constexpr int Hn  = 2048;
constexpr int Fn  = Dn * Cn;     // 1536
constexpr int IN1 = Fn + Ln;     // 1792 (GRU1 input width)
constexpr int Gn  = 3 * Hn;      // 6144 (gate width)

__device__ __forceinline__ unsigned short f32_to_bf16(float f) {
  unsigned int u = __float_as_uint(f);
  unsigned int lsb = (u >> 16) & 1u;
  u += 0x7fffu + lsb;            // round to nearest even
  return (unsigned short)(u >> 16);
}

union Frag16 { uint4 u4[2]; v16bf v; };

// A-operand fragment (16x32): lane = M row (lane&15); lanes>=16 take K+8
// halves. Per lane: bf16 K-chunks [kb..kb+7] and [kb+16..kb+23], kb = 0|8.
__device__ __forceinline__ v16bf load_fragA(const unsigned short* p) {
  Frag16 f;
  f.u4[0] = *reinterpret_cast<const uint4*>(p);
  f.u4[1] = *reinterpret_cast<const uint4*>(p + 16);
  return f.v;
}

// B-operand fragment (32x16): lane = N col (lane&15); per lane K rows
// [kb..kb+7] and [kb+16..kb+23], kb = 0|8 (same order as A), row stride ld.
__device__ __forceinline__ v16bf load_fragB(const unsigned short* p, size_t ld) {
  union { unsigned short u[16]; v16bf v; } f;
#pragma unroll
  for (int s = 0; s < 16; ++s) f.u[s] = p[(size_t)((s & 7) + ((s >> 3) << 4)) * ld];
  return f.v;
}

#define WMMA_BF16(ACC, Af, Bf) \
  ACC = __builtin_amdgcn_wmma_f32_16x16x32_bf16(false, (Af), false, (Bf), (short)0, ACC, false, false)

// Load one K-block's 6 fragments (2 A rows of 16, 4 N tiles of 16).
#define LOAD_SET(SA0, SA1, SB0, SB1, SB2, SB3) \
  SA0 = load_fragA(ap);                        \
  SA1 = load_fragA(ap + aRow16);               \
  SB0 = load_fragB(bp, (size_t)N);             \
  SB1 = load_fragB(bp + 16, (size_t)N);        \
  SB2 = load_fragB(bp + 32, (size_t)N);        \
  SB3 = load_fragB(bp + 48, (size_t)N)

// 8 WMMAs of one K-block against the 8 accumulators.
#define WMMA_SET(SA0, SA1, SB0, SB1, SB2, SB3) \
  WMMA_BF16(acc00, SA0, SB0);                  \
  WMMA_BF16(acc01, SA0, SB1);                  \
  WMMA_BF16(acc02, SA0, SB2);                  \
  WMMA_BF16(acc03, SA0, SB3);                  \
  WMMA_BF16(acc10, SA1, SB0);                  \
  WMMA_BF16(acc11, SA1, SB1);                  \
  WMMA_BF16(acc12, SA1, SB2);                  \
  WMMA_BF16(acc13, SA1, SB3)

// ---------------------------------------------------------------------------
// One-time: f32 (N,K) weight -> bf16 transposed (K,N) so the WMMA B operand
// reads N-contiguous rows, coalesced across lanes (one 16-bit load per K row).
// ---------------------------------------------------------------------------
__global__ void k_w_to_bf16_t(const float* __restrict__ W,
                              unsigned short* __restrict__ Wt,
                              int Nr, int K) {
  long long idx = (long long)blockIdx.x * blockDim.x + threadIdx.x;
  long long total = (long long)Nr * K;
  if (idx >= total) return;
  int k = (int)(idx / Nr);
  int n = (int)(idx % Nr);
  Wt[idx] = f32_to_bf16(W[(long long)n * K + k]);   // coalesced writes
}

// inp = concat(one_hot0, z) in bf16.  out0[:, F-1] = 1.
__global__ void k_init_inp(const float* __restrict__ z,
                           unsigned short* __restrict__ inp) {
  int idx = blockIdx.x * blockDim.x + threadIdx.x;
  if (idx >= Bn * IN1) return;
  int b = idx / IN1, c = idx % IN1;
  float v = (c < Fn) ? ((c == Fn - 1) ? 1.0f : 0.0f) : z[b * Ln + (c - Fn)];
  inp[idx] = f32_to_bf16(v);
}

// h1_0 = tanh(z @ W_init^T + b_init)   (one-time, tiny: 0.27 GFLOP)
__global__ void k_init_h1(const float* __restrict__ z,
                          const float* __restrict__ Wi,
                          const float* __restrict__ bi,
                          float* __restrict__ h1,
                          unsigned short* __restrict__ h1b) {
  int idx = blockIdx.x * blockDim.x + threadIdx.x;
  if (idx >= Bn * Hn) return;
  int b = idx / Hn, n = idx % Hn;
  const float* zr = z  + b * Ln;
  const float* wr = Wi + n * Ln;
  float acc = bi[n];
#pragma unroll 4
  for (int l = 0; l < Ln; ++l) acc += zr[l] * wr[l];
  float h = tanhf(acc);
  h1[idx]  = h;
  h1b[idx] = f32_to_bf16(h);
}

// ---------------------------------------------------------------------------
// Dual-problem GEMM: blocks [0, blocks0) compute problem 0, the rest
// problem 1 (the two gate GEMMs of a GRU cell are independent -> one launch).
// C(512, N) = A(512, K)bf16 @ Wt(K, N)bf16 + bias, f32 accumulate.
// Wave tile 32(M) x 64(N): 8 accumulators, 8 WMMA per K-block of 32.
// Manual 2x-unrolled ping-pong: while one fragment set feeds WMMAs the other
// set's loads are in flight (K/32 is even for both K=1792 and K=2048).
// ---------------------------------------------------------------------------
__global__ __launch_bounds__(256, 2)
void k_gemm_bf16(const unsigned short* __restrict__ A0,
                 const unsigned short* __restrict__ W0,
                 const float* __restrict__ bias0,
                 float* __restrict__ C0, int K0,
                 const unsigned short* __restrict__ A1,
                 const unsigned short* __restrict__ W1,
                 const float* __restrict__ bias1,
                 float* __restrict__ C1, int K1,
                 int N, int tilesN, int blocks0) {
  const int w    = threadIdx.x >> 5;
  const int lane = threadIdx.x & 31;

  const unsigned short* A;
  const unsigned short* Wt;
  const float* bias;
  float* C;
  int K, widx;
  if ((int)blockIdx.x < blocks0) {
    A = A0; Wt = W0; bias = bias0; C = C0; K = K0;
    widx = blockIdx.x * 8 + w;
  } else {
    A = A1; Wt = W1; bias = bias1; C = C1; K = K1;
    widx = (blockIdx.x - blocks0) * 8 + w;
  }
  const int tn = widx % tilesN;
  const int tm = widx / tilesN;          // 32-row M tile

  const int row  = lane & 15;
  const int kb   = (lane >> 4) << 3;                 // 0|8   (A K-half)
  const int krow = (lane & 15) + ((lane >> 4) << 4); // 0..31 (B K row)
  const int n0   = tn * 64;

  const unsigned short* ap = A  + (size_t)(tm * 32 + row) * K + kb;
  const size_t aRow16 = (size_t)16 * K;              // M+16 fragment offset
  const unsigned short* bp = Wt + (size_t)kb * N + n0 + row;
  const size_t bstep = (size_t)32 * N;

  v8f acc00 = {}, acc01 = {}, acc02 = {}, acc03 = {};
  v8f acc10 = {}, acc11 = {}, acc12 = {}, acc13 = {};

  v16bf xa0, xa1, xb0, xb1, xb2, xb3;   // set 0
  v16bf ya0, ya1, yb0, yb1, yb2, yb3;   // set 1

  // Prologue: set0 <- K-block 0.
  LOAD_SET(xa0, xa1, xb0, xb1, xb2, xb3);

  // Pairs of K-blocks; requires (K/32) even (true: 56, 64).
#pragma unroll 1
  for (int k = 32; k + 32 < K; k += 64) {
    ap += 32; bp += bstep;
    LOAD_SET(ya0, ya1, yb0, yb1, yb2, yb3);     // set1 <- block k
    WMMA_SET(xa0, xa1, xb0, xb1, xb2, xb3);     // compute block k-32
    ap += 32; bp += bstep;
    LOAD_SET(xa0, xa1, xb0, xb1, xb2, xb3);     // set0 <- block k+32
    WMMA_SET(ya0, ya1, yb0, yb1, yb2, yb3);     // compute block k
  }
  // Epilogue: set0 holds block K-64; load+compute block K-32.
  ap += 32; bp += bstep;
  LOAD_SET(ya0, ya1, yb0, yb1, yb2, yb3);
  WMMA_SET(xa0, xa1, xb0, xb1, xb2, xb3);
  WMMA_SET(ya0, ya1, yb0, yb1, yb2, yb3);

  // D layout: lane&15 = N col; lanes>=16 hold M+8; VGPR r = M row offset.
  const int nc = n0 + (lane & 15);
  const float bv0 = bias[nc];
  const float bv1 = bias[nc + 16];
  const float bv2 = bias[nc + 32];
  const float bv3 = bias[nc + 48];
  const int mB0 = tm * 32 + ((lane >> 4) << 3);
#pragma unroll
  for (int r = 0; r < 8; ++r) {
    float* crow = C + (size_t)(mB0 + r) * N + nc;
    crow[0]  = acc00[r] + bv0;
    crow[16] = acc01[r] + bv1;
    crow[32] = acc02[r] + bv2;
    crow[48] = acc03[r] + bv3;
  }
#pragma unroll
  for (int r = 0; r < 8; ++r) {
    float* crow = C + (size_t)(mB0 + 16 + r) * N + nc;
    crow[0]  = acc10[r] + bv0;
    crow[16] = acc11[r] + bv1;
    crow[32] = acc12[r] + bv2;
    crow[48] = acc13[r] + bv3;
  }
}

// ---------------------------------------------------------------------------
// GRU pointwise.  gi/gh are (B, 3H) = [r | z | n] chunks.
// ---------------------------------------------------------------------------
__device__ __forceinline__ float sigmoidf_(float x) {
  return 1.0f / (1.0f + __expf(-x));
}

__global__ void k_gru_pw1(const float* __restrict__ gi, const float* __restrict__ gh,
                          float* __restrict__ h1, unsigned short* __restrict__ h1b,
                          float* __restrict__ h2, unsigned short* __restrict__ h2b,
                          int step) {
  int idx = blockIdx.x * blockDim.x + threadIdx.x;
  if (idx >= Bn * Hn) return;
  int b = idx / Hn, j = idx % Hn;
  const float* gib = gi + (size_t)b * Gn;
  const float* ghb = gh + (size_t)b * Gn;
  float r  = sigmoidf_(gib[j]          + ghb[j]);
  float zg = sigmoidf_(gib[Hn + j]     + ghb[Hn + j]);
  float n  = tanhf    (gib[2 * Hn + j] + r * ghb[2 * Hn + j]);
  float hn = (1.0f - zg) * n + zg * h1[idx];
  h1[idx]  = hn;
  h1b[idx] = f32_to_bf16(hn);
  if (step == 0) {            // hx[1] initialized to hx[0] at step 0
    h2[idx]  = hn;
    h2b[idx] = f32_to_bf16(hn);
  }
}

__global__ void k_gru_pw2(const float* __restrict__ gi, const float* __restrict__ gh,
                          float* __restrict__ h2, unsigned short* __restrict__ h2b) {
  int idx = blockIdx.x * blockDim.x + threadIdx.x;
  if (idx >= Bn * Hn) return;
  int b = idx / Hn, j = idx % Hn;
  const float* gib = gi + (size_t)b * Gn;
  const float* ghb = gh + (size_t)b * Gn;
  float r  = sigmoidf_(gib[j]          + ghb[j]);
  float zg = sigmoidf_(gib[Hn + j]     + ghb[Hn + j]);
  float n  = tanhf    (gib[2 * Hn + j] + r * ghb[2 * Hn + j]);
  float hn = (1.0f - zg) * n + zg * h2[idx];
  h2[idx]  = hn;
  h2b[idx] = f32_to_bf16(hn);
}

// ---------------------------------------------------------------------------
// Per (b,d): log-softmax over C=12 classes (stride D in logits), write logp
// to d_out[b,t, c*D+d]; greedy one-hot (first max, strict >) into the bf16
// input buffer at layout d*C+c (contiguous per thread).
// ---------------------------------------------------------------------------
__global__ void k_softmax_onehot(const float* __restrict__ logits,
                                 float* __restrict__ out,
                                 unsigned short* __restrict__ inp,
                                 int t) {
  int idx = blockIdx.x * blockDim.x + threadIdx.x;
  if (idx >= Bn * Dn) return;
  int b = idx / Dn, d = idx % Dn;
  const float* lb = logits + (size_t)b * Fn;
  float x[Cn];
  float mx = -3.0e38f;
  int am = 0;
#pragma unroll
  for (int c = 0; c < Cn; ++c) {
    x[c] = lb[c * Dn + d];
    if (x[c] > mx) { mx = x[c]; am = c; }   // first-max semantics
  }
  float s = 0.0f;
#pragma unroll
  for (int c = 0; c < Cn; ++c) s += __expf(x[c] - mx);
  float lse = mx + __logf(s);
  float* ob = out + ((size_t)b * Tn + t) * Fn;
#pragma unroll
  for (int c = 0; c < Cn; ++c) ob[c * Dn + d] = x[c] - lse;  // coalesced in d
  unsigned short* ib = inp + (size_t)b * IN1 + d * Cn;
  const unsigned short one = 0x3f80, zero = 0;
#pragma unroll
  for (int c = 0; c < Cn; ++c) ib[c] = (c == am) ? one : zero;
}

// ---------------------------------------------------------------------------
// Host-side orchestration (stream-ordered, graph-capturable).
// ---------------------------------------------------------------------------
extern "C" void kernel_launch(void* const* d_in, const int* in_sizes, int n_in,
                              void* d_out, int out_size, void* d_ws, size_t ws_size,
                              hipStream_t stream) {
  const float* z      = (const float*)d_in[0];
  const float* W_ih1  = (const float*)d_in[1];
  const float* W_hh1  = (const float*)d_in[2];
  const float* b_ih1  = (const float*)d_in[3];
  const float* b_hh1  = (const float*)d_in[4];
  const float* W_ih2  = (const float*)d_in[5];
  const float* W_hh2  = (const float*)d_in[6];
  const float* b_ih2  = (const float*)d_in[7];
  const float* b_hh2  = (const float*)d_in[8];
  const float* W_init = (const float*)d_in[9];
  const float* b_init = (const float*)d_in[10];
  const float* W_out  = (const float*)d_in[11];
  const float* b_out  = (const float*)d_in[12];
  float* out = (float*)d_out;

  // Workspace layout (all offsets 256B-aligned by construction).
  char* ws = (char*)d_ws;
  size_t o = 0;
  auto take = [&](size_t bytes) { char* p = ws + o; o += (bytes + 255) & ~(size_t)255; return p; };
  unsigned short* Wih1t = (unsigned short*)take((size_t)IN1 * Gn * 2);  // (1792, 6144)
  unsigned short* Whh1t = (unsigned short*)take((size_t)Hn  * Gn * 2);  // (2048, 6144)
  unsigned short* Wih2t = (unsigned short*)take((size_t)Hn  * Gn * 2);
  unsigned short* Whh2t = (unsigned short*)take((size_t)Hn  * Gn * 2);
  unsigned short* Woutt = (unsigned short*)take((size_t)Hn  * Fn * 2);  // (2048, 1536)
  unsigned short* inpb  = (unsigned short*)take((size_t)Bn * IN1 * 2);
  float*          h1f   = (float*)take((size_t)Bn * Hn * 4);
  unsigned short* h1b   = (unsigned short*)take((size_t)Bn * Hn * 2);
  float*          h2f   = (float*)take((size_t)Bn * Hn * 4);
  unsigned short* h2b   = (unsigned short*)take((size_t)Bn * Hn * 2);
  float*          gi    = (float*)take((size_t)Bn * Gn * 4);
  float*          gh    = (float*)take((size_t)Bn * Gn * 4);
  float*          lgt   = (float*)take((size_t)Bn * Fn * 4);
  (void)ws_size; (void)n_in; (void)in_sizes; (void)out_size;

  const int TB = 256;
  auto blocks = [](long long n, int tb) { return (int)((n + tb - 1) / tb); };

  // One-time weight conversion + transpose to bf16 (L2-resident afterwards).
  k_w_to_bf16_t<<<blocks((long long)Gn * IN1, TB), TB, 0, stream>>>(W_ih1, Wih1t, Gn, IN1);
  k_w_to_bf16_t<<<blocks((long long)Gn * Hn,  TB), TB, 0, stream>>>(W_hh1, Whh1t, Gn, Hn);
  k_w_to_bf16_t<<<blocks((long long)Gn * Hn,  TB), TB, 0, stream>>>(W_ih2, Wih2t, Gn, Hn);
  k_w_to_bf16_t<<<blocks((long long)Gn * Hn,  TB), TB, 0, stream>>>(W_hh2, Whh2t, Gn, Hn);
  k_w_to_bf16_t<<<blocks((long long)Fn * Hn,  TB), TB, 0, stream>>>(W_out, Woutt, Fn, Hn);

  k_init_inp<<<blocks((long long)Bn * IN1, TB), TB, 0, stream>>>(z, inpb);
  k_init_h1 <<<blocks((long long)Bn * Hn,  TB), TB, 0, stream>>>(z, W_init, b_init, h1f, h1b);

  // GEMM grids: waves = (512/32) * (N/64); 8 waves (256 thr) per block.
  const int tnG   = Gn / 64;                        // 96
  const int tnF   = Fn / 64;                        // 24
  const int gridG = (Bn / 32) * tnG / 8;            // 192 blocks per gate GEMM
  const int gridF = (Bn / 32) * tnF / 8;            // 48 blocks (out proj)
  const int gridPW = blocks((long long)Bn * Hn, TB);
  const int gridSM = blocks((long long)Bn * Dn, TB);

  for (int t = 0; t < Tn; ++t) {
    // GRU cell 1: gi (from concat input) and gh (from h1) fused in one launch
    k_gemm_bf16<<<2 * gridG, TB, 0, stream>>>(
        inpb, Wih1t, b_ih1, gi, IN1,
        h1b,  Whh1t, b_hh1, gh, Hn,
        Gn, tnG, gridG);
    k_gru_pw1<<<gridPW, TB, 0, stream>>>(gi, gh, h1f, h1b, h2f, h2b, t);
    // GRU cell 2
    k_gemm_bf16<<<2 * gridG, TB, 0, stream>>>(
        h1b, Wih2t, b_ih2, gi, Hn,
        h2b, Whh2t, b_hh2, gh, Hn,
        Gn, tnG, gridG);
    k_gru_pw2<<<gridPW, TB, 0, stream>>>(gi, gh, h2f, h2b);
    // Output projection + log-softmax + greedy one-hot feedback
    k_gemm_bf16<<<gridF, TB, 0, stream>>>(
        h2b, Woutt, b_out, lgt, Hn,
        h2b, Woutt, b_out, lgt, Hn,   // second problem unused (blocks0 = grid)
        Fn, tnF, gridF);
    k_softmax_onehot<<<gridSM, TB, 0, stream>>>(lgt, out, inpb, t);
  }
}
